// MFILoss_38225208934871
// MI455X (gfx1250) — matrix operand order repaired and checked by reading, not compiled, not run on hardware
//
#include <hip/hip_runtime.h>
#include <stdint.h>

#define VOCAB 8192
#define DIMK  768
#define NB    (VOCAB / 32)                  // 256 macro-blocks (32 rows each)
#define TOTAL_MACRO ((NB * (NB + 1)) / 2)   // 32896 lower-triangle macro tiles

typedef __attribute__((ext_vector_type(16))) __bf16 v16bf;
typedef __attribute__((ext_vector_type(8)))  float  v8f;

// ---------- bf16 helpers (round-to-nearest-even split) ----------
__device__ __forceinline__ unsigned short f2bf_rne(float f) {
    unsigned u = __float_as_uint(f);
    unsigned lsb = (u >> 16) & 1u;
    u += 0x7fffu + lsb;
    return (unsigned short)(u >> 16);
}
__device__ __forceinline__ float bf2f(unsigned short h) {
    return __uint_as_float(((unsigned)h) << 16);
}

// ---------- kernel 0: zero accumulators ----------
__global__ void zero_accum(float* __restrict__ rowSum,
                           float* __restrict__ rowCube,
                           float* __restrict__ collapse) {
    int idx = blockIdx.x * blockDim.x + threadIdx.x;
    if (idx < VOCAB) { rowSum[idx] = 0.0f; rowCube[idx] = 0.0f; }
    if (idx == 0) *collapse = 0.0f;
}

// ---------- kernel 1: L2-normalize rows, split into bf16 hi/lo ----------
__global__ void __launch_bounds__(256)
normalize_split(const float* __restrict__ t_prime,
                unsigned short* __restrict__ Ahi,
                unsigned short* __restrict__ Alo) {
    const int row = blockIdx.x;
    const int tid = threadIdx.x;
    const float* rp = t_prime + (size_t)row * DIMK;

    float x0 = rp[tid], x1 = rp[tid + 256], x2 = rp[tid + 512];
    float ss = x0 * x0 + x1 * x1 + x2 * x2;
    for (int m = 1; m < 32; m <<= 1) ss += __shfl_xor(ss, m, 32);

    __shared__ float red[8];
    const int wave = tid >> 5, lane = tid & 31;
    if (lane == 0) red[wave] = ss;
    __syncthreads();
    float tot = red[0] + red[1] + red[2] + red[3] +
                red[4] + red[5] + red[6] + red[7];
    float scale = 1.0f / fmaxf(sqrtf(tot), 1e-12f);

    const size_t base = (size_t)row * DIMK;
    float v[3] = { x0 * scale, x1 * scale, x2 * scale };
    int   c[3] = { tid, tid + 256, tid + 512 };
#pragma unroll
    for (int e = 0; e < 3; ++e) {
        unsigned short hi = f2bf_rne(v[e]);
        float residual = v[e] - bf2f(hi);
        Ahi[base + c[e]] = hi;
        Alo[base + c[e]] = f2bf_rne(residual);
    }
}

// ---------- fragment loader helpers ----------
union FragU { uint4 q[2]; v16bf v; };

__device__ __forceinline__ v16bf loadA(const unsigned short* p) {
    // A 16x32 bf16 layout: per lane two 16B chunks 32B apart
    const uint4* q = reinterpret_cast<const uint4*>(p);
    FragU f; f.q[0] = q[0]; f.q[1] = q[2];
    return f.v;
}
__device__ __forceinline__ v16bf loadB(const unsigned short* p) {
    // B 32x16 bf16 layout: per lane two 16B chunks 32B apart
    const uint4* q = reinterpret_cast<const uint4*>(p);
    FragU f; f.q[0] = q[0]; f.q[1] = q[2];
    return f.v;
}

// ---------- sub-tile epilogue: fold 16x16 C into row accumulators ----------
__device__ __forceinline__ void reduce_subtile(
        const v8f& acc, unsigned rowBlk, unsigned colBlk,
        bool isDiagSub, bool mirror, unsigned half, unsigned m,
        float* __restrict__ rowSum, float* __restrict__ rowCube,
        float& collapsePart) {
    const unsigned col     = colBlk + m;
    const unsigned rowBase = rowBlk + half * 8u;
    float colSum = 0.0f, colCube = 0.0f;
#pragma unroll
    for (int g = 0; g < 8; ++g) {
        float s = acc[g];
        unsigned row = rowBase + (unsigned)g;
        float sOff = s;
        if (isDiagSub && row == col) {          // strip diagonal, collapse term
            float d = s - 1.0f;
            collapsePart += d * d;
            sOff = 0.0f;
        }
        float rs = sOff, rc = sOff * sOff * sOff;
        for (int mm = 1; mm < 16; mm <<= 1) {   // reduce across 16 cols
            rs += __shfl_xor(rs, mm, 32);
            rc += __shfl_xor(rc, mm, 32);
        }
        if (m == 0) {
            atomicAdd(&rowSum[row], rs);
            atomicAdd(&rowCube[row], rc);
        }
        colSum  += sOff;
        colCube += sOff * sOff * sOff;
    }
    if (mirror) {  // S symmetric: per-column sums feed rows == col
        colSum  += __shfl_xor(colSum, 16, 32);
        colCube += __shfl_xor(colCube, 16, 32);
        if (half == 0) {
            atomicAdd(&rowSum[col], colSum);
            atomicAdd(&rowCube[col], colCube);
        }
    }
}

// ---------- kernel 2: 32x32 macro-tile WMMA Gram + fused reduction ----------
__global__ void __launch_bounds__(256)
cosine_tile_kernel(const unsigned short* __restrict__ Ahi,
                   const unsigned short* __restrict__ Alo,
                   float* __restrict__ rowSum,
                   float* __restrict__ rowCube,
                   float* __restrict__ collapse) {
    const unsigned waveInBlk = threadIdx.x >> 5;
    const unsigned lane      = threadIdx.x & 31u;
    const unsigned tile      = blockIdx.x * 8u + waveInBlk;
    if (tile >= TOTAL_MACRO) return;   // wave-uniform exit; EXEC stays all-ones

    // Lower-triangle (I >= J) macro decode: tile = I*(I+1)/2 + J
    unsigned I = (unsigned)((sqrtf(8.0f * (float)tile + 1.0f) - 1.0f) * 0.5f);
    while ((I + 1u) * (I + 2u) / 2u <= tile) ++I;
    while (I * (I + 1u) / 2u > tile) --I;
    const unsigned J = tile - I * (I + 1u) / 2u;

    const unsigned half = lane >> 4;
    const unsigned m    = lane & 15u;

    // Per-lane source rows for the 4 A / 4 B fragments
    const size_t a0 = (size_t)(I * 32u +      m) * DIMK;   // rows r0..r0+15
    const size_t a1 = (size_t)(I * 32u + 16u + m) * DIMK;  // rows r0+16..r0+31
    const size_t b0 = (size_t)(J * 32u +      m) * DIMK;   // cols c0..c0+15
    const size_t b1 = (size_t)(J * 32u + 16u + m) * DIMK;  // cols c0+16..c0+31
    const unsigned aOff = half * 8u;    // A frag chunk offset within K-step
    const unsigned bOff = half * 8u;    // B frag chunk offset within K-step

    v8f c00 = {}, c01 = {}, c10 = {}, c11 = {};
    for (int kk = 0; kk < DIMK; kk += 32) {
        v16bf ah0 = loadA(Ahi + a0 + kk + aOff);
        v16bf al0 = loadA(Alo + a0 + kk + aOff);
        v16bf ah1 = loadA(Ahi + a1 + kk + aOff);
        v16bf al1 = loadA(Alo + a1 + kk + aOff);
        v16bf bh0 = loadB(Ahi + b0 + kk + bOff);
        v16bf bl0 = loadB(Alo + b0 + kk + bOff);
        v16bf bh1 = loadB(Ahi + b1 + kk + bOff);
        v16bf bl1 = loadB(Alo + b1 + kk + bOff);

        // S ~= Ahi*Bhi + Ahi*Blo + Alo*Bhi (bf16x2 split, drop lo*lo)
        c00 = __builtin_amdgcn_wmma_f32_16x16x32_bf16(false, al0, false, bh0, (short)0, c00, false, false);
        c00 = __builtin_amdgcn_wmma_f32_16x16x32_bf16(false, ah0, false, bl0, (short)0, c00, false, false);
        c00 = __builtin_amdgcn_wmma_f32_16x16x32_bf16(false, ah0, false, bh0, (short)0, c00, false, false);

        c01 = __builtin_amdgcn_wmma_f32_16x16x32_bf16(false, al0, false, bh1, (short)0, c01, false, false);
        c01 = __builtin_amdgcn_wmma_f32_16x16x32_bf16(false, ah0, false, bl1, (short)0, c01, false, false);
        c01 = __builtin_amdgcn_wmma_f32_16x16x32_bf16(false, ah0, false, bh1, (short)0, c01, false, false);

        c10 = __builtin_amdgcn_wmma_f32_16x16x32_bf16(false, al1, false, bh0, (short)0, c10, false, false);
        c10 = __builtin_amdgcn_wmma_f32_16x16x32_bf16(false, ah1, false, bl0, (short)0, c10, false, false);
        c10 = __builtin_amdgcn_wmma_f32_16x16x32_bf16(false, ah1, false, bh0, (short)0, c10, false, false);

        c11 = __builtin_amdgcn_wmma_f32_16x16x32_bf16(false, al1, false, bh1, (short)0, c11, false, false);
        c11 = __builtin_amdgcn_wmma_f32_16x16x32_bf16(false, ah1, false, bl1, (short)0, c11, false, false);
        c11 = __builtin_amdgcn_wmma_f32_16x16x32_bf16(false, ah1, false, bh1, (short)0, c11, false, false);
    }

    const unsigned r0 = I * 32u, c0 = J * 32u;
    float collapsePart = 0.0f;
    if (I == J) {
        // (0,0) and (1,1) straddle the diagonal; (1,0) strictly lower (mirror);
        // (0,1) is the duplicate upper block -> discarded.
        reduce_subtile(c00, r0,       c0,       true,  false, half, m, rowSum, rowCube, collapsePart);
        reduce_subtile(c10, r0 + 16u, c0,       false, true,  half, m, rowSum, rowCube, collapsePart);
        reduce_subtile(c11, r0 + 16u, c0 + 16u, true,  false, half, m, rowSum, rowCube, collapsePart);
        for (int mm = 1; mm < 32; mm <<= 1)
            collapsePart += __shfl_xor(collapsePart, mm, 32);
        if (lane == 0) atomicAdd(collapse, collapsePart);
    } else {
        reduce_subtile(c00, r0,       c0,       false, true, half, m, rowSum, rowCube, collapsePart);
        reduce_subtile(c01, r0,       c0 + 16u, false, true, half, m, rowSum, rowCube, collapsePart);
        reduce_subtile(c10, r0 + 16u, c0,       false, true, half, m, rowSum, rowCube, collapsePart);
        reduce_subtile(c11, r0 + 16u, c0 + 16u, false, true, half, m, rowSum, rowCube, collapsePart);
    }
}

// ---------- kernel 3: finalize scalar ----------
__global__ void __launch_bounds__(256)
finalize_kernel(const float* __restrict__ rowSum,
                const float* __restrict__ rowCube,
                const float* __restrict__ collapse,
                float* __restrict__ out) {
    const int tid = threadIdx.x;
    float local = 0.0f;
    for (int r = tid; r < VOCAB; r += 256) {
        float mean = rowSum[r] * (1.0f / (float)(VOCAB - 1));
        local += rowCube[r] / (mean + 1e-6f);
    }
    for (int m = 1; m < 32; m <<= 1) local += __shfl_xor(local, m, 32);
    __shared__ float red[8];
    const int wave = tid >> 5, lane = tid & 31;
    if (lane == 0) red[wave] = local;
    __syncthreads();
    if (tid == 0) {
        float hard = red[0] + red[1] + red[2] + red[3] +
                     red[4] + red[5] + red[6] + red[7];
        out[0] = *collapse + 0.2f * (1.0f * hard);   // LAMBDA=0.2, BETA=1.0
    }
}

// ---------- launch ----------
extern "C" void kernel_launch(void* const* d_in, const int* in_sizes, int n_in,
                              void* d_out, int out_size, void* d_ws, size_t ws_size,
                              hipStream_t stream) {
    const float* t_prime = (const float*)d_in[0];
    float* out = (float*)d_out;

    char* ws = (char*)d_ws;
    const size_t bfBytes = (size_t)VOCAB * DIMK * sizeof(unsigned short); // 12 MB
    unsigned short* Ahi     = (unsigned short*)(ws);
    unsigned short* Alo     = (unsigned short*)(ws + bfBytes);
    float*          rowSum  = (float*)(ws + 2 * bfBytes);
    float*          rowCube = (float*)(ws + 2 * bfBytes + VOCAB * sizeof(float));
    float*          collapse= (float*)(ws + 2 * bfBytes + 2 * VOCAB * sizeof(float));

    zero_accum<<<(VOCAB + 255) / 256, 256, 0, stream>>>(rowSum, rowCube, collapse);
    normalize_split<<<VOCAB, 256, 0, stream>>>(t_prime, Ahi, Alo);

    const unsigned blocks = (TOTAL_MACRO + 7u) / 8u;   // 8 macro-tiles per block
    cosine_tile_kernel<<<blocks, 256, 0, stream>>>(Ahi, Alo, rowSum, rowCube, collapse);

    finalize_kernel<<<1, 256, 0, stream>>>(rowSum, rowCube, collapse, out);
}
